// GNN_61100204752973
// MI455X (gfx1250) — compile-verified, matrix operand order checked by reading, not run on hardware
//
#include <hip/hip_runtime.h>
#include <hip/hip_bf16.h>

typedef __attribute__((ext_vector_type(16))) _Float16 v16h;
typedef __attribute__((ext_vector_type(8)))  _Float16 v8h;
typedef __attribute__((ext_vector_type(8)))  float    v8f;
typedef __attribute__((ext_vector_type(4)))  float    v4f;

// ---------------- problem constants ----------------
static constexpr int cB   = 256;
static constexpr int cN   = 192;
static constexpr int cDIN = 67;
static constexpr int cDM  = 512;
static constexpr int cNH  = 8;
static constexpr int cDH  = 64;
static constexpr int cDFF = 2048;
static constexpr int cGH  = 2;
static constexpr int cGC  = 16;
static constexpr int cNC  = 1000;
static constexpr int cBN  = cB * cN;            // 49152 rows

// ---------------- generic batched WMMA GEMM ----------------
// D[z] = relu?( alpha * A[z] * op(B[z]) + bias + add[z] + pe )
// A: [M,K] (f32 or f16), B: btrans ? [N,K] : [K,N], C: [M,N] (f32 or f16)
struct GP {
  const void*  A;
  const float* B;
  void*        C;
  const float* add;
  const float* bias;
  const float* pe;
  long long sAo, sAi, sBo, sBi, sCo, sCi, sAddo, sAddi;   // element strides
  int inner, batch;
  int M, N, K, lda, ldb, ldc, ldadd, peMod, ldpe;
  float alpha;
  int btrans, relu, c_f16;
  int avec, bvec;                 // 128-bit load eligibility of A / B operand
};

template <typename AT>
__global__ __launch_bounds__(256) void gemm_wmma(GP p) {
  constexpr int BM = 128, BN = 64, BK = 32, PAD = 8;
  __shared__ _Float16 As[2][BM][BK + PAD];
  __shared__ _Float16 Bs[2][BN][BK + PAD];

  const int z  = blockIdx.z;
  const int zo = z / p.inner, zi = z % p.inner;
  const AT*    Ap = (const AT*)p.A + (long long)zo * p.sAo + (long long)zi * p.sAi;
  const float* Bp = p.B + (long long)zo * p.sBo + (long long)zi * p.sBi;
  const long long cOff   = (long long)zo * p.sCo  + (long long)zi * p.sCi;
  const long long addOff = (long long)zo * p.sAddo + (long long)zi * p.sAddi;

  const int bm0 = blockIdx.y * BM, bn0 = blockIdx.x * BN;
  const int tid  = threadIdx.x;
  const int lane = tid & 31, wid = tid >> 5;
  const int wm = (wid & 3) * 32;      // 4 waves along M
  const int wn = (wid >> 2) * 32;     // 2 waves along N
  const int r  = lane & 15, kh = lane >> 4;

  // per-thread staging coordinates
  const int aRowL  = tid >> 1;               // A: 2 threads per row, 16 elems each
  const int aKbase = (tid & 1) * 16;
  const int aGrow  = bm0 + aRowL;
  const bool aRowOk = (aGrow < p.M);
  const AT* aRowPtr = Ap + (long long)aGrow * p.lda;

  const int bNL    = tid >> 2;               // B: 4 threads per row, 8 elems each
  const int bKbase = (tid & 3) * 8;
  const int bGn    = bn0 + bNL;
  const bool bColOk = (bGn < p.N);

  // ---- tile staging (global -> f16 LDS), fast interior path + guarded edges ----
  auto stage = [&](int k0, int buf) {
    const bool kFull = (k0 + BK <= p.K);
    // A tile (128 x 32)
    if (kFull & aRowOk & (p.avec != 0)) {
      v8h h0, h1;
      if constexpr (sizeof(AT) == 2) {
        const v8h* s = (const v8h*)(aRowPtr + k0 + aKbase);
        h0 = s[0]; h1 = s[1];
      } else {
        const v4f* s = (const v4f*)((const float*)aRowPtr + k0 + aKbase);
        v4f f0 = s[0], f1 = s[1], f2 = s[2], f3 = s[3];
#pragma unroll
        for (int i = 0; i < 4; ++i) { h0[i] = (_Float16)f0[i]; h0[4 + i] = (_Float16)f1[i]; }
#pragma unroll
        for (int i = 0; i < 4; ++i) { h1[i] = (_Float16)f2[i]; h1[4 + i] = (_Float16)f3[i]; }
      }
      v8h* dst = (v8h*)&As[buf][aRowL][aKbase];
      dst[0] = h0; dst[1] = h1;
    } else if (kFull & aRowOk) {
      _Float16 tmp[16];
#pragma unroll
      for (int i = 0; i < 16; ++i) tmp[i] = (_Float16)(float)aRowPtr[k0 + aKbase + i];
#pragma unroll
      for (int i = 0; i < 16; ++i) As[buf][aRowL][aKbase + i] = tmp[i];
    } else {
#pragma unroll
      for (int i = 0; i < 16; ++i) {
        const int gk = k0 + aKbase + i;
        float v = 0.f;
        if (aRowOk && gk < p.K) v = (float)aRowPtr[gk];
        As[buf][aRowL][aKbase + i] = (_Float16)v;
      }
    }
    // B tile (64 x 32)
    if (kFull & bColOk & (p.btrans != 0) & (p.bvec != 0)) {
      const v4f* s = (const v4f*)(Bp + (long long)bGn * p.ldb + k0 + bKbase);
      v4f f0 = s[0], f1 = s[1];
      v8h h0;
#pragma unroll
      for (int i = 0; i < 4; ++i) { h0[i] = (_Float16)f0[i]; h0[4 + i] = (_Float16)f1[i]; }
      *(v8h*)&Bs[buf][bNL][bKbase] = h0;
    } else if (kFull & bColOk & (p.btrans != 0)) {
      _Float16 tmp[8];
      const float* row = Bp + (long long)bGn * p.ldb + k0 + bKbase;
#pragma unroll
      for (int i = 0; i < 8; ++i) tmp[i] = (_Float16)row[i];
#pragma unroll
      for (int i = 0; i < 8; ++i) Bs[buf][bNL][bKbase + i] = tmp[i];
    } else if (kFull & bColOk) {
      _Float16 tmp[8];
#pragma unroll
      for (int i = 0; i < 8; ++i)
        tmp[i] = (_Float16)Bp[(long long)(k0 + bKbase + i) * p.ldb + bGn];
#pragma unroll
      for (int i = 0; i < 8; ++i) Bs[buf][bNL][bKbase + i] = tmp[i];
    } else {
#pragma unroll
      for (int i = 0; i < 8; ++i) {
        const int gk = k0 + bKbase + i;
        float v = 0.f;
        if (bColOk && gk < p.K)
          v = p.btrans ? Bp[(long long)bGn * p.ldb + gk]
                       : Bp[(long long)gk * p.ldb + bGn];
        Bs[buf][bNL][bKbase + i] = (_Float16)v;
      }
    }
  };

  v8f zero = {};
  v8f acc[2][2];
  acc[0][0] = zero; acc[0][1] = zero; acc[1][0] = zero; acc[1][1] = zero;

  // ---- software-pipelined main loop: stage k+1 while computing k ----
  stage(0, 0);
  int buf = 0;
  for (int k0 = 0; k0 < p.K; k0 += BK) {
    __syncthreads();                       // tile[buf] visible to all waves
    if (k0 + BK < p.K) stage(k0 + BK, buf ^ 1);

    // fragments per documented 16-bit A/B layout
    v16h af[2], bf[2];
#pragma unroll
    for (int mi = 0; mi < 2; ++mi) {
      const int row = wm + mi * 16 + r;
#pragma unroll
      for (int pq = 0; pq < 8; ++pq) {
        const int kofs = ((pq < 4) ? 2 * pq : 16 + 2 * (pq - 4)) + kh * 8;
        af[mi][2 * pq]     = As[buf][row][kofs];
        af[mi][2 * pq + 1] = As[buf][row][kofs + 1];
      }
    }
#pragma unroll
    for (int ni = 0; ni < 2; ++ni) {
      const int col = wn + ni * 16 + r;
#pragma unroll
      for (int pq = 0; pq < 8; ++pq) {
        const int kofs = ((pq < 4) ? 2 * pq : 16 + 2 * (pq - 4)) + kh * 8;
        bf[ni][2 * pq]     = Bs[buf][col][kofs];
        bf[ni][2 * pq + 1] = Bs[buf][col][kofs + 1];
      }
    }
#pragma unroll
    for (int mi = 0; mi < 2; ++mi)
#pragma unroll
      for (int ni = 0; ni < 2; ++ni)
        acc[mi][ni] = __builtin_amdgcn_wmma_f32_16x16x32_f16(
            false, af[mi], false, bf[ni], (short)0, acc[mi][ni], false, false);
    buf ^= 1;
  }

  // ---- epilogue: C/D layout (VGPR e: M = e + 8*lanehalf, N = lane&15) ----
  float*    Cf = (float*)p.C + cOff;
  _Float16* Ch = (_Float16*)p.C + cOff;
  const bool interior = (bm0 + BM <= p.M) && (bn0 + BN <= p.N);
#pragma unroll
  for (int mi = 0; mi < 2; ++mi)
#pragma unroll
    for (int ni = 0; ni < 2; ++ni)
#pragma unroll
      for (int e = 0; e < 8; ++e) {
        const int row = bm0 + wm + mi * 16 + kh * 8 + e;
        const int col = bn0 + wn + ni * 16 + r;
        if (interior || (row < p.M && col < p.N)) {
          float v = p.alpha * acc[mi][ni][e];
          if (p.bias) v += p.bias[col];
          if (p.add)  v += p.add[addOff + (long long)row * p.ldadd + col];
          if (p.pe)   v += p.pe[(long long)(row % p.peMod) * p.ldpe + col];
          if (p.relu) v = fmaxf(v, 0.f);
          if (p.c_f16) Ch[(long long)row * p.ldc + col] = (_Float16)v;
          else         Cf[(long long)row * p.ldc + col] = v;
        }
      }
}

// ---------------- positional encoding ----------------
__global__ void pe_kernel(float* pe) {
  int idx = blockIdx.x * blockDim.x + threadIdx.x;        // 192*256 pairs
  if (idx >= cN * (cDM / 2)) return;
  int pos = idx / (cDM / 2);
  int i   = idx % (cDM / 2);
  float div = __expf((2.f * i) * (-logf(10000.f) / (float)cDM));
  float ang = pos * div;
  pe[pos * cDM + 2 * i]     = __sinf(ang);
  pe[pos * cDM + 2 * i + 1] = __cosf(ang);
}

// ---------------- LayerNorm over 512, wave per row ----------------
__global__ void ln_kernel(const float* __restrict__ src, float* __restrict__ dst,
                          const float* __restrict__ g, const float* __restrict__ be,
                          int rows) {
  int row = blockIdx.x * 8 + (threadIdx.x >> 5);
  int lane = threadIdx.x & 31;
  if (row >= rows) return;
  const float* x = src + (long long)row * cDM;
  float s = 0.f;
  for (int i = lane; i < cDM; i += 32) s += x[i];
  for (int m = 16; m; m >>= 1) s += __shfl_xor(s, m, 32);
  float mean = s * (1.f / cDM);
  float v = 0.f;
  for (int i = lane; i < cDM; i += 32) { float d = x[i] - mean; v += d * d; }
  for (int m = 16; m; m >>= 1) v += __shfl_xor(v, m, 32);
  float inv = rsqrtf(v * (1.f / cDM) + 1e-5f);
  float* o = dst + (long long)row * cDM;
  for (int i = lane; i < cDM; i += 32) o[i] = (x[i] - mean) * inv * g[i] + be[i];
}

// ---------------- softmax over 192 keys, wave per row, in place ----------------
__global__ void softmax192(float* S, int rows) {
  int row = blockIdx.x * 8 + (threadIdx.x >> 5);
  int lane = threadIdx.x & 31;
  if (row >= rows) return;
  float* x = S + (long long)row * cN;
  float m = -1e30f;
  for (int i = lane; i < cN; i += 32) m = fmaxf(m, x[i]);
  for (int k = 16; k; k >>= 1) m = fmaxf(m, __shfl_xor(m, k, 32));
  float s = 0.f;
  for (int i = lane; i < cN; i += 32) { float e = __expf(x[i] - m); x[i] = e; s += e; }
  for (int k = 16; k; k >>= 1) s += __shfl_xor(s, k, 32);
  float inv = 1.f / s;
  for (int i = lane; i < cN; i += 32) x[i] *= inv;
}

// ---------------- madj symmetrize / outputs ----------------
__global__ void sym_kernel(const float* __restrict__ madj, const float* __restrict__ adj,
                           float* __restrict__ nadjOut, float* __restrict__ asOut) {
  long long idx = (long long)blockIdx.x * blockDim.x + threadIdx.x;
  if (idx >= (long long)cB * cN * cN) return;
  int b   = (int)(idx / (cN * cN));
  int rem = (int)(idx % (cN * cN));
  int i = rem / cN, j = rem % cN;
  float m1 = madj[idx];
  float m2 = madj[(long long)b * cN * cN + (long long)j * cN + i];
  float sy = 0.5f * (m1 + m2);
  float as = 0.5f * (m1 - m2);
  nadjOut[idx] = fmaxf(adj[idx] + sy, 0.f);
  asOut[idx]   = as;
}

// ---------------- GAT e_s / e_d ----------------
__global__ void esed_kernel(const float* __restrict__ xp, const float* __restrict__ a_s,
                            const float* __restrict__ a_d, float* __restrict__ es,
                            float* __restrict__ ed) {
  int id = blockIdx.x * blockDim.x + threadIdx.x;
  if (id >= cBN) return;
  const float* xr = xp + (long long)id * (cGH * cGC);
#pragma unroll
  for (int h = 0; h < cGH; ++h) {
    float s1 = 0.f, s2 = 0.f;
#pragma unroll
    for (int c = 0; c < cGC; ++c) {
      float v = xr[h * cGC + c];
      s1 += v * a_s[h * cGC + c];
      s2 += v * a_d[h * cGC + c];
    }
    es[id * cGH + h] = s1;
    ed[id * cGH + h] = s2;
  }
}

// ---------------- GAT masked-softmax aggregation, block per (b, dst) ----------------
__global__ __launch_bounds__(64) void gat_agg(const float* __restrict__ xp,
                                              const float* __restrict__ es,
                                              const float* __restrict__ ed,
                                              const float* __restrict__ nadj,
                                              const float* __restrict__ bias,
                                              float* __restrict__ out) {
  __shared__ float sAdj[cN];
  __shared__ float sEs[cN * cGH];
  __shared__ float sEd[cGH];
  int bt = blockIdx.x;
  int b = bt / cN, t = bt % cN;
  int tid = threadIdx.x;
  for (int s = tid; s < cN; s += 64)
    sAdj[s] = nadj[((long long)b * cN + s) * cN + t];
  for (int i = tid; i < cN * cGH; i += 64)
    sEs[i] = es[(long long)b * cN * cGH + i];
  if (tid < cGH) sEd[tid] = ed[((long long)b * cN + t) * cGH + tid];
  __syncthreads();
  if (tid < cGH * cGC) {
    int h = tid / cGC, c = tid % cGC;
    float edv = sEd[h];
    float m = -1e30f;
    for (int s = 0; s < cN; ++s) {
      if (sAdj[s] > 0.f || s == t) {
        float e = sEs[s * cGH + h] + edv;
        e = (e > 0.f) ? e : 0.2f * e;       // leaky relu 0.2
        m = fmaxf(m, e);
      }
    }
    float den = 0.f, accv = 0.f;
    for (int s = 0; s < cN; ++s) {
      if (sAdj[s] > 0.f || s == t) {
        float e = sEs[s * cGH + h] + edv;
        e = (e > 0.f) ? e : 0.2f * e;
        float w = __expf(e - m);
        den  += w;
        accv += w * xp[(((long long)b * cN + s) * cGH + h) * cGC + c];
      }
    }
    float o = accv / den + bias[h * cGC + c];
    out[(((long long)b * cN + t) * cGH + h) * cGC + c] = fmaxf(o, 0.f);
  }
}

// ---------------- host-side helpers ----------------
static inline GP base_gp() {
  GP p = {};
  p.inner = 1; p.batch = 1; p.alpha = 1.f;
  p.peMod = 1; p.ldpe = 0; p.ldadd = 0;
  return p;
}

static void launch_gemm(hipStream_t st, GP p, bool a_is_f16) {
  // 128-bit staging eligibility (strides divisible by 128b element count)
  const int aElems = a_is_f16 ? 8 : 4;
  p.avec = (p.lda % aElems == 0) ? 1 : 0;
  p.bvec = (p.btrans && (p.ldb % 4 == 0)) ? 1 : 0;
  dim3 g((p.N + 63) / 64, (p.M + 127) / 128, p.batch);
  dim3 blk(256);
  if (a_is_f16) gemm_wmma<_Float16><<<g, blk, 0, st>>>(p);
  else          gemm_wmma<float><<<g, blk, 0, st>>>(p);
}

extern "C" void kernel_launch(void* const* d_in, const int* in_sizes, int n_in,
                              void* d_out, int out_size, void* d_ws, size_t ws_size,
                              hipStream_t stream) {
  (void)in_sizes; (void)n_in; (void)out_size; (void)ws_size;
  const float* x     = (const float*)d_in[0];
  const float* adj   = (const float*)d_in[1];
  /* indices d_in[2] unused */
  const float* W_up  = (const float*)d_in[3];
  const float* b_up  = (const float*)d_in[4];
  const float* W_lin = (const float*)d_in[5];
  const float* b_lin = (const float*)d_in[6];
  const float* Wq    = (const float*)d_in[7];
  const float* bq    = (const float*)d_in[8];
  const float* Wk    = (const float*)d_in[9];
  const float* bk    = (const float*)d_in[10];
  const float* Wv    = (const float*)d_in[11];
  const float* bv    = (const float*)d_in[12];
  const float* Wo    = (const float*)d_in[13];
  const float* bo    = (const float*)d_in[14];
  const float* g1    = (const float*)d_in[15];
  const float* be1   = (const float*)d_in[16];
  const float* W1    = (const float*)d_in[17];
  const float* b1    = (const float*)d_in[18];
  const float* W2    = (const float*)d_in[19];
  const float* b2    = (const float*)d_in[20];
  const float* g2    = (const float*)d_in[21];
  const float* be2   = (const float*)d_in[22];
  const float* W_map = (const float*)d_in[23];
  const float* b_map = (const float*)d_in[24];
  const float* Wg1 = (const float*)d_in[25]; const float* as1 = (const float*)d_in[26];
  const float* ad1 = (const float*)d_in[27]; const float* bg1 = (const float*)d_in[28];
  const float* Wg2 = (const float*)d_in[29]; const float* as2 = (const float*)d_in[30];
  const float* ad2 = (const float*)d_in[31]; const float* bg2 = (const float*)d_in[32];
  const float* Wg3 = (const float*)d_in[33]; const float* as3 = (const float*)d_in[34];
  const float* ad3 = (const float*)d_in[35]; const float* bg3 = (const float*)d_in[36];
  const float* Wcls = (const float*)d_in[37]; const float* bcls = (const float*)d_in[38];

  float* out  = (float*)d_out;
  float* yOut    = out;                                  // [256,1000]
  float* nadjOut = out + (long long)cB * cNC;            // [256,192,192]
  float* asOut   = nadjOut + (long long)cB * cN * cN;    // [256,192,192]

  // ---------------- workspace layout (float offsets) ----------------
  float* wsf = (float*)d_ws;
  const long long SZ_HD = (long long)cBN * cDM;          // 25165824
  const long long OFF_PE = 0;
  const long long OFF_H  = OFF_PE + (long long)cN * cDM; // 98304
  const long long OFF_T0 = OFF_H + SZ_HD;
  const long long OFF_Q  = OFF_T0 + SZ_HD;
  const long long OFF_K  = OFF_Q + SZ_HD;
  const long long OFF_V  = OFF_K + SZ_HD;
  const long long OFF_S  = OFF_V + SZ_HD;                // 75.5MB attention chunk / h0 / madj
  // GAT buffers alias V (free after attention):
  const long long OFF_XP  = OFF_V;
  const long long OFF_XGA = OFF_V + (long long)cBN * 32;
  const long long OFF_XGB = OFF_V + (long long)cBN * 64;
  const long long OFF_ES  = OFF_V + (long long)cBN * 96;
  const long long OFF_ED  = OFF_ES + (long long)cBN * 2;

  // 0) positional encoding table
  pe_kernel<<<(cN * (cDM / 2) + 255) / 256, 256, 0, stream>>>(wsf + OFF_PE);

  // 1) h0 = adj + x @ W_up^T + b_up          [49152,192]
  {
    GP p = base_gp();
    p.A = x; p.lda = cDIN; p.B = W_up; p.ldb = cDIN; p.btrans = 1;
    p.M = cBN; p.N = cN; p.K = cDIN;
    p.bias = b_up; p.add = adj; p.ldadd = cN;
    p.C = wsf + OFF_S; p.ldc = cN;
    launch_gemm(stream, p, false);
  }
  // 2) h = h0 @ W_lin^T + b_lin + PE         [49152,512]
  {
    GP p = base_gp();
    p.A = wsf + OFF_S; p.lda = cN; p.B = W_lin; p.ldb = cN; p.btrans = 1;
    p.M = cBN; p.N = cDM; p.K = cN;
    p.bias = b_lin; p.pe = wsf + OFF_PE; p.peMod = cN; p.ldpe = cDM;
    p.C = wsf + OFF_H; p.ldc = cDM;
    launch_gemm(stream, p, false);
  }
  // 3) q,k,v projections
  const float* Wqkv[3] = {Wq, Wk, Wv};
  const float* bqkv[3] = {bq, bk, bv};
  const long long offqkv[3] = {OFF_Q, OFF_K, OFF_V};
  for (int i = 0; i < 3; ++i) {
    GP p = base_gp();
    p.A = wsf + OFF_H; p.lda = cDM; p.B = Wqkv[i]; p.ldb = cDM; p.btrans = 1;
    p.M = cBN; p.N = cDM; p.K = cDM; p.bias = bqkv[i];
    p.C = wsf + offqkv[i]; p.ldc = cDM;
    launch_gemm(stream, p, false);
  }
  // 4) attention, chunked over batch (CBCH=64 -> S fits 75.5MB)
  {
    const int CBCH = 64, NCHUNK = cB / CBCH;
    for (int c = 0; c < NCHUNK; ++c) {
      const long long co = (long long)c * CBCH * cN * cDM;
      const int zb = CBCH * cNH;                         // 512 batched gemms
      // S = 0.125 * q @ k^T  per (b,h): M=N=192, K=64
      {
        GP p = base_gp();
        p.batch = zb; p.inner = cNH;
        p.A = wsf + OFF_Q + co; p.lda = cDM; p.sAo = (long long)cN * cDM; p.sAi = cDH;
        p.B = wsf + OFF_K + co; p.ldb = cDM; p.sBo = (long long)cN * cDM; p.sBi = cDH;
        p.btrans = 1;
        p.C = wsf + OFF_S; p.ldc = cN;
        p.sCo = (long long)cNH * cN * cN; p.sCi = (long long)cN * cN;
        p.M = cN; p.N = cN; p.K = cDH; p.alpha = 0.125f;
        launch_gemm(stream, p, false);
      }
      // row softmax (in place)
      {
        int rows = zb * cN;                              // 98304
        softmax192<<<(rows + 7) / 8, 256, 0, stream>>>(wsf + OFF_S, rows);
      }
      // O = S @ v  per (b,h): M=192, K=192, N=64 ; scatter into t0 at head cols
      {
        GP p = base_gp();
        p.batch = zb; p.inner = cNH;
        p.A = wsf + OFF_S; p.lda = cN;
        p.sAo = (long long)cNH * cN * cN; p.sAi = (long long)cN * cN;
        p.B = wsf + OFF_V + co; p.ldb = cDM; p.sBo = (long long)cN * cDM; p.sBi = cDH;
        p.btrans = 0;                                    // B is [K=j, N=d]
        p.C = wsf + OFF_T0 + co; p.ldc = cDM;
        p.sCo = (long long)cN * cDM; p.sCi = cDH;
        p.M = cN; p.N = cDH; p.K = cN;
        launch_gemm(stream, p, false);
      }
    }
  }
  // 5) o_proj + residual: t0 @ Wo^T + bo + h -> Q region
  {
    GP p = base_gp();
    p.A = wsf + OFF_T0; p.lda = cDM; p.B = Wo; p.ldb = cDM; p.btrans = 1;
    p.M = cBN; p.N = cDM; p.K = cDM; p.bias = bo;
    p.add = wsf + OFF_H; p.ldadd = cDM;
    p.C = wsf + OFF_Q; p.ldc = cDM;
    launch_gemm(stream, p, false);
  }
  // 6) LN1 -> h
  ln_kernel<<<(cBN + 7) / 8, 256, 0, stream>>>(wsf + OFF_Q, wsf + OFF_H, g1, be1, cBN);
  // 7) FFN1: relu(h @ W1^T + b1) -> f16 buffer over Q+K regions
  {
    GP p = base_gp();
    p.A = wsf + OFF_H; p.lda = cDM; p.B = W1; p.ldb = cDM; p.btrans = 1;
    p.M = cBN; p.N = cDFF; p.K = cDM; p.bias = b1; p.relu = 1; p.c_f16 = 1;
    p.C = wsf + OFF_Q; p.ldc = cDFF;
    launch_gemm(stream, p, false);
  }
  // 8) FFN2: ff1 @ W2^T + b2 + h -> t0
  {
    GP p = base_gp();
    p.A = wsf + OFF_Q; p.lda = cDFF; p.B = W2; p.ldb = cDFF; p.btrans = 1;
    p.M = cBN; p.N = cDM; p.K = cDFF; p.bias = b2;
    p.add = wsf + OFF_H; p.ldadd = cDM;
    p.C = wsf + OFF_T0; p.ldc = cDM;
    launch_gemm(stream, p, true);                        // A is f16
  }
  // 9) LN2 -> h
  ln_kernel<<<(cBN + 7) / 8, 256, 0, stream>>>(wsf + OFF_T0, wsf + OFF_H, g2, be2, cBN);
  // 10) madj = h @ W_map^T + b_map -> S region
  {
    GP p = base_gp();
    p.A = wsf + OFF_H; p.lda = cDM; p.B = W_map; p.ldb = cDM; p.btrans = 1;
    p.M = cBN; p.N = cN; p.K = cDM; p.bias = b_map;
    p.C = wsf + OFF_S; p.ldc = cN;
    launch_gemm(stream, p, false);
  }
  // 11) symmetrize: write nadj and madj_AS outputs
  {
    long long tot = (long long)cB * cN * cN;
    sym_kernel<<<(unsigned)((tot + 255) / 256), 256, 0, stream>>>(
        wsf + OFF_S, adj, nadjOut, asOut);
  }
  // 12) three GAT layers
  const float* Wg[3] = {Wg1, Wg2, Wg3};
  const float* aS[3] = {as1, as2, as3};
  const float* aD[3] = {ad1, ad2, ad3};
  const float* bG[3] = {bg1, bg2, bg3};
  const long long gin[3]  = {-1, OFF_XGA, OFF_XGB};      // -1 -> use x
  const long long gout[3] = {OFF_XGA, OFF_XGB, OFF_XGA};
  for (int l = 0; l < 3; ++l) {
    // xp = in @ Wg^T
    {
      GP p = base_gp();
      if (l == 0) { p.A = x; p.lda = cDIN; p.K = cDIN; }
      else        { p.A = wsf + gin[l]; p.lda = cGH * cGC; p.K = cGH * cGC; }
      p.B = Wg[l]; p.ldb = p.K; p.btrans = 1;
      p.M = cBN; p.N = cGH * cGC;
      p.C = wsf + OFF_XP; p.ldc = cGH * cGC;
      launch_gemm(stream, p, false);
    }
    esed_kernel<<<(cBN + 255) / 256, 256, 0, stream>>>(
        wsf + OFF_XP, aS[l], aD[l], wsf + OFF_ES, wsf + OFF_ED);
    gat_agg<<<cBN, 64, 0, stream>>>(wsf + OFF_XP, wsf + OFF_ES, wsf + OFF_ED,
                                    nadjOut, bG[l], wsf + gout[l]);
  }
  // 13) classifier: y = xg.reshape(B,6144) @ Wcls^T + bcls
  {
    GP p = base_gp();
    p.A = wsf + OFF_XGA; p.lda = cN * cGH * cGC;         // 6144
    p.B = Wcls; p.ldb = cN * cGH * cGC; p.btrans = 1;
    p.M = cB; p.N = cNC; p.K = cN * cGH * cGC;
    p.bias = bcls; p.C = yOut; p.ldc = cNC;
    launch_gemm(stream, p, false);
  }
}
